// MultiHeadSelfAttention_5239860101318
// MI455X (gfx1250) — matrix operand order repaired and checked by reading, not compiled, not run on hardware
//
#include <hip/hip_runtime.h>

#define D_MODEL 1024
#define N_SEQ   2048
#define N_BATCH 2
#define N_HEADS 16
#define D_K     64
#define M_ROWS  (N_BATCH * N_SEQ)   // 4096

typedef __attribute__((ext_vector_type(16))) __bf16 v16bf;
typedef __attribute__((ext_vector_type(8)))  float  v8f;

union Frag { unsigned int u[8]; v16bf v; };

__device__ __forceinline__ unsigned short f2bf(float f) {
  return __builtin_bit_cast(unsigned short, (__bf16)f);
}
__device__ __forceinline__ unsigned int pack2(float lo, float hi) {
  return (unsigned int)f2bf(lo) | ((unsigned int)f2bf(hi) << 16);
}
__device__ __forceinline__ v8f zero8() {
  v8f z = {0.f, 0.f, 0.f, 0.f, 0.f, 0.f, 0.f, 0.f};
  return z;
}
__device__ __forceinline__ v8f wmma_bf16(const Frag& a, const Frag& b, v8f c) {
  return __builtin_amdgcn_wmma_f32_16x16x32_bf16(false, a.v, false, b.v,
                                                 (short)0, c, false, false);
}

// xor-shuffle within 16-lane groups via ds_swizzle (group-of-32 mode:
// and_mask=0x1f, or=0, xor=m). Immediate must be a literal -> macro.
#define SWIZ(x, m) \
  __builtin_bit_cast(float, __builtin_amdgcn_ds_swizzle( \
      __builtin_bit_cast(int, (x)), (((m) << 10) | 0x1f)))

// --------------------------- TDM support -----------------------------------
#if defined(__gfx1250__) && \
    __has_builtin(__builtin_amdgcn_tensor_load_to_lds) && \
    __has_builtin(__builtin_amdgcn_s_wait_tensorcnt)
#define USE_TDM 1
#endif

#ifdef USE_TDM
typedef __attribute__((ext_vector_type(4))) unsigned int v4u;
typedef __attribute__((ext_vector_type(8))) int v8i;
typedef __attribute__((ext_vector_type(4))) int v4i;

__device__ __forceinline__ unsigned lds_offset_of(const void* p) {
  return (unsigned)(unsigned long long)
      (__attribute__((address_space(3))) const void*)p;
}

// 2D tile load: `rows` x `row_elems` of 2-byte elements, contiguous rows
// (tensor == tile, stride = row_elems), global -> LDS at byte offset lds_off.
// D# layout per CDNA5 ISA ch.8: group0 = {ctl, lds_addr, gaddr_lo, gaddr_hi|type},
// group1 = {data_size, dims/strides/tile dims}, groups 2/3 zero (2D tensor).
__device__ __forceinline__ void tdm_load_2d_bf16(const void* gaddr,
                                                 unsigned lds_off,
                                                 unsigned rows,
                                                 unsigned row_elems) {
  unsigned long long ga = (unsigned long long)gaddr;
  v4u g0;
  g0[0] = 1u;                                             // count=1, user mode
  g0[1] = lds_off;                                        // lds_addr (bytes)
  g0[2] = (unsigned)(ga & 0xffffffffu);                   // global_addr[31:0]
  g0[3] = (unsigned)((ga >> 32) & 0x01ffffffu)            // global_addr[56:32]
          | (2u << 30);                                   // type=2 (image)
  v8i g1;
  g1[0] = (int)(1u << 16);              // workgroup_mask=0, data_size=1 (2B)
  g1[1] = (int)(row_elems << 16);       // tensor_dim0[15:0]
  g1[2] = (int)(rows << 16);            // tensor_dim0[31:16]=0 | tensor_dim1[15:0]
  g1[3] = (int)(row_elems << 16);       // tensor_dim1[31:16]=0 | tile_dim0
  g1[4] = (int)rows;                    // tile_dim1 | tile_dim2=0
  g1[5] = (int)row_elems;               // tensor_dim0_stride[31:0]
  g1[6] = 0;                            // stride hi | tensor_dim1_stride lo
  g1[7] = 0;
  v4i z4 = {0, 0, 0, 0};
  v8i z8 = {0, 0, 0, 0, 0, 0, 0, 0};
  __builtin_amdgcn_tensor_load_to_lds(g0, g1, z4, z4, z8, 0);
}
#endif

// ---------------------------------------------------------------------------
// Kernel 1: QKV projections. out[b,h,n,dk] (bf16) = x @ W + bias, per z = q/k/v
// ---------------------------------------------------------------------------
__global__ __launch_bounds__(128)
void qkv_proj_kernel(const float* __restrict__ x,
                     const float* __restrict__ Wq, const float* __restrict__ bq,
                     const float* __restrict__ Wk, const float* __restrict__ bk,
                     const float* __restrict__ Wv, const float* __restrict__ bv,
                     unsigned short* __restrict__ qkv) {
  __shared__ __align__(16) unsigned short Ws[64 * 32];   // [col][k] transposed

  const int s = blockIdx.z;
  const float* W    = (s == 0) ? Wq : (s == 1) ? Wk : Wv;
  const float* bias = (s == 0) ? bq : (s == 1) ? bk : bv;
  unsigned short* outb = qkv + (size_t)s * ((size_t)M_ROWS * D_MODEL);

  const int r0   = blockIdx.x * 64;
  const int c0   = blockIdx.y * 64;
  const int tid  = threadIdx.x;
  const int wv   = tid >> 5;
  const int lane = tid & 31;
  const int g    = lane >> 4;
  const int ln   = lane & 15;
  const int arow = r0 + wv * 16 + ln;

  v8f acc[4];
  #pragma unroll
  for (int t = 0; t < 4; ++t) acc[t] = zero8();

  const unsigned int* WsU = (const unsigned int*)Ws;

  for (int k0 = 0; k0 < D_MODEL; k0 += 32) {
    __syncthreads();
    // stage W[k0..k0+31][c0..c0+63] -> Ws[cc][kk]: batch all global loads
    // (float4) before converts/stores so they overlap.
    float4 tmp[4];
    #pragma unroll
    for (int i = 0; i < 4; ++i) {
      int e4 = tid + i * 128;            // 512 float4s = 2048 floats
      int kk = e4 >> 4;
      int cc = (e4 & 15) * 4;
      tmp[i] = *(const float4*)&W[(size_t)(k0 + kk) * D_MODEL + c0 + cc];
    }
    #pragma unroll
    for (int i = 0; i < 4; ++i) {
      int e4 = tid + i * 128;
      int kk = e4 >> 4;
      int cc = (e4 & 15) * 4;
      Ws[(cc + 0) * 32 + kk] = f2bf(tmp[i].x);
      Ws[(cc + 1) * 32 + kk] = f2bf(tmp[i].y);
      Ws[(cc + 2) * 32 + kk] = f2bf(tmp[i].z);
      Ws[(cc + 3) * 32 + kk] = f2bf(tmp[i].w);
    }
    if (k0 + 32 < D_MODEL)
      __builtin_prefetch(&W[(size_t)(k0 + 32) * D_MODEL + c0 + (tid & 63)], 0, 0);
    __syncthreads();

    // A fragment straight from global fp32, converted to bf16 pairs
    Frag a;
    #pragma unroll
    for (int v = 0; v < 8; ++v) {
      int kk = g * 16 + 2 * v;
      const float* p = &x[(size_t)arow * D_MODEL + k0 + kk];
      a.u[v] = pack2(p[0], p[1]);
    }
    #pragma unroll
    for (int ct = 0; ct < 4; ++ct) {
      Frag b;
      #pragma unroll
      for (int v = 0; v < 8; ++v) {
        int cc = ct * 16 + ln;
        int kk = g * 16 + 2 * v;
        b.u[v] = WsU[(cc * 32 + kk) >> 1];
      }
      acc[ct] = wmma_bf16(a, b, acc[ct]);
    }
  }

  // epilogue: bias add, convert, scatter to [b,h,n,dk]
  #pragma unroll
  for (int ct = 0; ct < 4; ++ct) {
    int c = c0 + ct * 16 + ln;
    float bb = bias[c];
    #pragma unroll
    for (int i = 0; i < 8; ++i) {
      int r = r0 + wv * 16 + i + 8 * g;
      int b_ = r >> 11, n_ = r & 2047;
      int h_ = c >> 6,  dk_ = c & 63;
      outb[(((size_t)(b_ * N_HEADS + h_) * N_SEQ + n_) << 6) + dk_] =
          f2bf(acc[ct][i] + bb);
    }
  }
}

// ---------------------------------------------------------------------------
// Kernel 2: flash attention per (b,h). Each wave: 16 query rows, stream keys.
// K/V blocks staged to LDS by the Tensor Data Mover, double-buffered.
// ---------------------------------------------------------------------------
__global__ __launch_bounds__(128)
void attention_kernel(const unsigned short* __restrict__ Qb,
                      const unsigned short* __restrict__ Kb,
                      const unsigned short* __restrict__ Vb,
                      unsigned short* __restrict__ Ob) {
  __shared__ __align__(16) unsigned short Ks[2][32 * 64];  // [buf][key][d]
  __shared__ __align__(16) unsigned short Vs[2][32 * 64];  // [buf][key][d]
  __shared__ __align__(16) unsigned short Pl[4][16 * 32];  // per-wave P patch

  const int bh   = blockIdx.y;
  const size_t base = (size_t)bh * N_SEQ * D_K;
  const int tid  = threadIdx.x;
  const int wv   = tid >> 5;
  const int lane = tid & 31;
  const int g    = lane >> 4;
  const int ln   = lane & 15;
  const int q0   = blockIdx.x * 64 + wv * 16;

  // Q fragments for this wave's 16 rows (held in registers whole kernel)
  Frag aQ[2];
  #pragma unroll
  for (int ks = 0; ks < 2; ++ks)
    #pragma unroll
    for (int v = 0; v < 8; ++v) {
      int kk = ks * 32 + g * 16 + 2 * v;
      aQ[ks].u[v] =
          *(const unsigned int*)&Qb[base + (size_t)(q0 + ln) * D_K + kk];
    }

  float m8[8], l8[8];
  v8f accO[4];
  #pragma unroll
  for (int i = 0; i < 8; ++i) { m8[i] = -1e30f; l8[i] = 0.f; }
  #pragma unroll
  for (int t = 0; t < 4; ++t) accO[t] = zero8();

  const unsigned int* PlU = (const unsigned int*)Pl[wv];
  unsigned short* Pw = Pl[wv];

  // ---- prologue: stage key block 0 into buffer 0 ----
#ifdef USE_TDM
  if (wv == 0) {
    tdm_load_2d_bf16(&Kb[base], lds_offset_of(&Ks[0][0]), 32, 64);
    tdm_load_2d_bf16(&Vb[base], lds_offset_of(&Vs[0][0]), 32, 64);
  }
#else
  {
    const uint4* Ksrc = (const uint4*)&Kb[base];
    const uint4* Vsrc = (const uint4*)&Vb[base];
    #pragma unroll
    for (int i = 0; i < 2; ++i) {
      ((uint4*)Ks[0])[tid + i * 128] = Ksrc[tid + i * 128];
      ((uint4*)Vs[0])[tid + i * 128] = Vsrc[tid + i * 128];
    }
  }
#endif

  for (int j0 = 0; j0 < N_SEQ; j0 += 32) {
    const int buf = (j0 >> 5) & 1;
    const bool has_next = (j0 + 32) < N_SEQ;

#ifdef USE_TDM
    // issue next block's DMA into the other buffer, then wait for this block.
    // TDM ops complete in order per wave: tensorcnt<=2 => this block done.
    if (wv == 0) {
      if (has_next) {
        tdm_load_2d_bf16(&Kb[base + (size_t)(j0 + 32) * D_K],
                         lds_offset_of(&Ks[buf ^ 1][0]), 32, 64);
        tdm_load_2d_bf16(&Vb[base + (size_t)(j0 + 32) * D_K],
                         lds_offset_of(&Vs[buf ^ 1][0]), 32, 64);
        __builtin_amdgcn_s_wait_tensorcnt((short)2);
      } else {
        __builtin_amdgcn_s_wait_tensorcnt((short)0);
      }
    }
#endif
    __syncthreads();   // staged block `buf` visible to all waves

    const unsigned int* KsU = (const unsigned int*)Ks[buf];
    const unsigned short* Vn = Vs[buf];

    // S = (Q K^T) * 1/sqrt(dk), two 16-key subtiles
    v8f st[2];
    #pragma unroll
    for (int kt = 0; kt < 2; ++kt) {
      st[kt] = zero8();
      #pragma unroll
      for (int ks = 0; ks < 2; ++ks) {
        Frag b;
        #pragma unroll
        for (int v = 0; v < 8; ++v) {
          int key = kt * 16 + ln;
          int dd  = ks * 32 + g * 16 + 2 * v;
          b.u[v] = KsU[(key * 64 + dd) >> 1];
        }
        st[kt] = wmma_bf16(aQ[ks], b, st[kt]);
      }
      #pragma unroll
      for (int i = 0; i < 8; ++i) st[kt][i] *= 0.125f;
    }

    // online softmax: row reductions across 16-lane column groups
    float rm[8];
    #pragma unroll
    for (int i = 0; i < 8; ++i) {
      rm[i] = fmaxf(st[0][i], st[1][i]);
      rm[i] = fmaxf(rm[i], SWIZ(rm[i], 1));
      rm[i] = fmaxf(rm[i], SWIZ(rm[i], 2));
      rm[i] = fmaxf(rm[i], SWIZ(rm[i], 4));
      rm[i] = fmaxf(rm[i], SWIZ(rm[i], 8));
    }
    float alpha[8];
    #pragma unroll
    for (int i = 0; i < 8; ++i) {
      float mn = fmaxf(m8[i], rm[i]);
      alpha[i] = __expf(m8[i] - mn);
      m8[i] = mn;
    }
    #pragma unroll
    for (int kt = 0; kt < 2; ++kt)
      #pragma unroll
      for (int i = 0; i < 8; ++i) st[kt][i] = __expf(st[kt][i] - m8[i]);
    float rs[8];
    #pragma unroll
    for (int i = 0; i < 8; ++i) {
      rs[i] = st[0][i] + st[1][i];
      rs[i] += SWIZ(rs[i], 1);
      rs[i] += SWIZ(rs[i], 2);
      rs[i] += SWIZ(rs[i], 4);
      rs[i] += SWIZ(rs[i], 8);
      l8[i] = l8[i] * alpha[i] + rs[i];
    }
    #pragma unroll
    for (int t = 0; t < 4; ++t)
      #pragma unroll
      for (int i = 0; i < 8; ++i) accO[t][i] *= alpha[i];

    // P: C-layout fp32 -> per-wave LDS -> bf16 A-fragment (intra-wave, in-order)
    #pragma unroll
    for (int kt = 0; kt < 2; ++kt)
      #pragma unroll
      for (int i = 0; i < 8; ++i) {
        int row = i + 8 * g;
        Pw[row * 32 + kt * 16 + ln] = f2bf(st[kt][i]);
      }
    Frag aP;
    #pragma unroll
    for (int v = 0; v < 8; ++v) {
      int kk = g * 16 + 2 * v;
      aP.u[v] = PlU[(ln * 32 + kk) >> 1];
    }
    // O += P @ V  (V natural [key][d]: gather key-pairs as two u16 loads)
    #pragma unroll
    for (int ct = 0; ct < 4; ++ct) {
      Frag b;
      #pragma unroll
      for (int v = 0; v < 8; ++v) {
        int dk = ct * 16 + ln;
        int k0 = g * 16 + 2 * v;
        b.u[v] = (unsigned int)Vn[(k0 + 0) * 64 + dk]
               | ((unsigned int)Vn[(k0 + 1) * 64 + dk] << 16);
      }
      accO[ct] = wmma_bf16(aP, b, accO[ct]);
    }

    __syncthreads();   // everyone done reading buf before it is re-staged

#ifndef USE_TDM
    if (has_next) {
      const uint4* Ksrc = (const uint4*)&Kb[base + (size_t)(j0 + 32) * D_K];
      const uint4* Vsrc = (const uint4*)&Vb[base + (size_t)(j0 + 32) * D_K];
      #pragma unroll
      for (int i = 0; i < 2; ++i) {
        ((uint4*)Ks[buf ^ 1])[tid + i * 128] = Ksrc[tid + i * 128];
        ((uint4*)Vs[buf ^ 1])[tid + i * 128] = Vsrc[tid + i * 128];
      }
    }
#endif
  }

  // epilogue: normalize, store to [b, n, h*64+dk] bf16 for output projection
  const int b_ = bh >> 4, h_ = bh & 15;
  #pragma unroll
  for (int i = 0; i < 8; ++i) {
    float inv = 1.0f / l8[i];
    int n_ = q0 + i + 8 * g;
    #pragma unroll
    for (int ct = 0; ct < 4; ++ct) {
      int dk = ct * 16 + ln;
      Ob[(size_t)(b_ * N_SEQ + n_) * D_MODEL + h_ * D_K + dk] =
          f2bf(accO[ct][i] * inv);
    }
  }
}

// ---------------------------------------------------------------------------
// Kernel 3: output projection. out (fp32) = Ob(bf16) @ Wo + bo
// ---------------------------------------------------------------------------
__global__ __launch_bounds__(128)
void out_proj_kernel(const unsigned short* __restrict__ Ob,
                     const float* __restrict__ Wo, const float* __restrict__ bo,
                     float* __restrict__ out) {
  __shared__ __align__(16) unsigned short Ws[64 * 32];

  const int r0   = blockIdx.x * 64;
  const int c0   = blockIdx.y * 64;
  const int tid  = threadIdx.x;
  const int wv   = tid >> 5;
  const int lane = tid & 31;
  const int g    = lane >> 4;
  const int ln   = lane & 15;
  const int arow = r0 + wv * 16 + ln;

  v8f acc[4];
  #pragma unroll
  for (int t = 0; t < 4; ++t) acc[t] = zero8();

  const unsigned int* WsU = (const unsigned int*)Ws;

  for (int k0 = 0; k0 < D_MODEL; k0 += 32) {
    __syncthreads();
    float4 tmp[4];
    #pragma unroll
    for (int i = 0; i < 4; ++i) {
      int e4 = tid + i * 128;
      int kk = e4 >> 4;
      int cc = (e4 & 15) * 4;
      tmp[i] = *(const float4*)&Wo[(size_t)(k0 + kk) * D_MODEL + c0 + cc];
    }
    #pragma unroll
    for (int i = 0; i < 4; ++i) {
      int e4 = tid + i * 128;
      int kk = e4 >> 4;
      int cc = (e4 & 15) * 4;
      Ws[(cc + 0) * 32 + kk] = f2bf(tmp[i].x);
      Ws[(cc + 1) * 32 + kk] = f2bf(tmp[i].y);
      Ws[(cc + 2) * 32 + kk] = f2bf(tmp[i].z);
      Ws[(cc + 3) * 32 + kk] = f2bf(tmp[i].w);
    }
    if (k0 + 32 < D_MODEL)
      __builtin_prefetch(&Wo[(size_t)(k0 + 32) * D_MODEL + c0 + (tid & 63)], 0, 0);
    __syncthreads();

    Frag a;
    #pragma unroll
    for (int v = 0; v < 8; ++v) {
      int kk = g * 16 + 2 * v;
      a.u[v] = *(const unsigned int*)&Ob[(size_t)arow * D_MODEL + k0 + kk];
    }
    #pragma unroll
    for (int ct = 0; ct < 4; ++ct) {
      Frag b;
      #pragma unroll
      for (int v = 0; v < 8; ++v) {
        int cc = ct * 16 + ln;
        int kk = g * 16 + 2 * v;
        b.u[v] = WsU[(cc * 32 + kk) >> 1];
      }
      acc[ct] = wmma_bf16(a, b, acc[ct]);
    }
  }

  #pragma unroll
  for (int ct = 0; ct < 4; ++ct) {
    int c = c0 + ct * 16 + ln;
    float bb = bo[c];
    #pragma unroll
    for (int i = 0; i < 8; ++i) {
      int r = r0 + wv * 16 + i + 8 * g;
      out[(size_t)r * D_MODEL + c] = acc[ct][i] + bb;
    }
  }
}

// ---------------------------------------------------------------------------
extern "C" void kernel_launch(void* const* d_in, const int* in_sizes, int n_in,
                              void* d_out, int out_size, void* d_ws, size_t ws_size,
                              hipStream_t stream) {
  const float* x  = (const float*)d_in[0];
  const float* Wq = (const float*)d_in[1];
  const float* bq = (const float*)d_in[2];
  const float* Wk = (const float*)d_in[3];
  const float* bk = (const float*)d_in[4];
  const float* Wv = (const float*)d_in[5];
  const float* bv = (const float*)d_in[6];
  const float* Wo = (const float*)d_in[7];
  const float* bo = (const float*)d_in[8];
  float* out = (float*)d_out;

  const size_t mat = (size_t)M_ROWS * D_MODEL;   // 4M bf16 elements each
  unsigned short* qkv = (unsigned short*)d_ws;
  unsigned short* Qb = qkv;
  unsigned short* Kb = qkv + mat;
  unsigned short* Vb = qkv + 2 * mat;
  unsigned short* Ob = qkv + 3 * mat;

  qkv_proj_kernel<<<dim3(M_ROWS / 64, D_MODEL / 64, 3), 128, 0, stream>>>(
      x, Wq, bq, Wk, bk, Wv, bv, qkv);
  attention_kernel<<<dim3(N_SEQ / 64, N_BATCH * N_HEADS), 128, 0, stream>>>(
      Qb, Kb, Vb, Ob);
  out_proj_kernel<<<dim3(M_ROWS / 64, D_MODEL / 64), 128, 0, stream>>>(
      Ob, Wo, bo, out);
}
